// Attention_76493367542388
// MI455X (gfx1250) — matrix operand order repaired and checked by reading, not compiled, not run on hardware
//
#include <hip/hip_runtime.h>
#include <hip/hip_bf16.h>

typedef __attribute__((ext_vector_type(16))) __bf16 v16bf;
typedef __attribute__((ext_vector_type(8)))  __bf16 v8bf;
typedef __attribute__((ext_vector_type(8)))  float  v8f;

union F8   { v8f  v; float  f[8]; };
union BF8  { v8bf v; __bf16 e[8]; };
union BF16x16 { v16bf v; v8bf h[2]; };

#define WMMA_BF16(a, b, c) \
  __builtin_amdgcn_wmma_f32_16x16x32_bf16(false, (a), false, (b), (short)0, (c), false, false)

#define AS3 __attribute__((address_space(3)))

// raw 32-bit LDS byte offset of a generic pointer to __shared__ memory
__device__ __forceinline__ unsigned lds_addr(const void* p) {
  return (unsigned)(unsigned long)(AS3 const void*)p;
}

// async DMA: 16 bytes global -> LDS, tracked by ASYNCcnt (no VGPR data path)
__device__ __forceinline__ void async_ld_b128(unsigned lds_off, const void* g) {
  unsigned long long ga = (unsigned long long)(size_t)g;
  asm volatile("global_load_async_to_lds_b128 %0, %1, off"
               :: "v"(lds_off), "v"(ga) : "memory");
}

__device__ __forceinline__ void wait_async0() {
  asm volatile("s_wait_asynccnt 0x0" ::: "memory");
}

// ---------------------------------------------------------------------------
// WMMA fragment loaders (layouts per CDNA5 ISA 7.12.2, wave32)
// A 16x32 bf16: lane l (l<16) = row l, K = 0..7 & 16..23 ; lanes 16..31 same
// rows, K = 8..15 & 24..31.  base points at (row0, k0), ld in elements.
__device__ __forceinline__ v16bf load_frag_a(const __bf16* base, int ld, int lane) {
  const int r = lane & 15;
  const int g = lane >> 4;
  const __bf16* p = base + (size_t)r * ld + g * 8;
  BF16x16 u;
  u.h[0] = *(const v8bf*)(p);
  u.h[1] = *(const v8bf*)(p + 16);
  return u.v;
}

// B 32x16 bf16: lane l = column (l&15); lanes 0..15 hold K=0..7 & 16..23, lanes
// 16..31 hold K=8..15 & 24..31 (same K order as A).  Memory layout expected:
// element (k, n) at base[n*ld + k]  (i.e. n-major, contiguous in k).
__device__ __forceinline__ v16bf load_frag_b(const __bf16* base, int ld, int lane) {
  const int n = lane & 15;
  const int g = lane >> 4;
  const __bf16* p = base + (size_t)n * ld + g * 8;
  BF16x16 u;
  u.h[0] = *(const v8bf*)(p);
  u.h[1] = *(const v8bf*)(p + 16);
  return u.v;
}

// ---------------------------------------------------------------------------
// fp32 -> bf16 conversion (8 elems/thread)
__global__ void cvt_f32_bf16(const float* __restrict__ in, __bf16* __restrict__ out, int n) {
  int i = (blockIdx.x * blockDim.x + threadIdx.x) * 8;
  if (i >= n) return;
  float4 a = *(const float4*)(in + i);
  float4 b = *(const float4*)(in + i + 4);
  BF8 r;
  r.e[0] = (__bf16)a.x; r.e[1] = (__bf16)a.y; r.e[2] = (__bf16)a.z; r.e[3] = (__bf16)a.w;
  r.e[4] = (__bf16)b.x; r.e[5] = (__bf16)b.y; r.e[6] = (__bf16)b.z; r.e[7] = (__bf16)b.w;
  *(v8bf*)(out + i) = r.v;
}

// ---------------------------------------------------------------------------
// bf16 GEMM, 128x128 block tile, K-step 64, 8 waves (4x2), wave tile 32x64.
// Double-buffered pipeline: A tile DMA'd via GLOBAL_LOAD_ASYNC_TO_LDS into
// the alternate buffer while the current tile is consumed by WMMA; B tile
// register-prefetched then transposed into LDS so B-fragments are
// k-contiguous ds_load_b128 pairs.
// mode 0: C = x @ Wqkv (N=3072); epilogue applies RoPE and scatters
//         Q[B,H,S,64] (scaled by 1/8), K[B,H,S,64], Vt[B,H,64,S] (bf16).
// mode 1: plain fp32 store to Cf[M,N].
__global__ __launch_bounds__(256) void gemm_bf16_kernel(
    const __bf16* __restrict__ A, const __bf16* __restrict__ B,
    int M, int N, int K, int mode,
    float* __restrict__ Cf,
    __bf16* __restrict__ Qb, __bf16* __restrict__ Kb, __bf16* __restrict__ Vtb)
{
  __shared__ __align__(16) __bf16 Asm[2][128 * 64];   // row-major [m][k], 2x16KB
  __shared__ __align__(16) __bf16 Bsm[2][128 * 64];   // transposed [n][k], 2x16KB

  const int tid  = threadIdx.x;
  const int lane = tid & 31;
  const int wid  = tid >> 5;
  const int wm   = wid >> 1;      // 0..3
  const int wn   = wid & 1;       // 0..1
  const int m0   = blockIdx.x * 128;
  const int n0   = blockIdx.y * 128;

  const unsigned a_base = lds_addr(&Asm[0][0]);

  const int bk   = tid >> 2;          // 0..63
  const int boff = (tid & 3) * 32;    // 0..96

  // A tile: 128x64 = 1024 x 16B chunks; LDS offset linear in chunk index.
  const int ci0  = tid * 4;
  const int arow = ci0 >> 2;          // one row per thread, 4 segs
  auto stage_a = [&](int k0, int p) {
#pragma unroll
    for (int c = 0; c < 4; c++) {
      int ci = ci0 + c;               // row = arow, seg = c
      async_ld_b128(a_base + (unsigned)p * (128 * 64 * 2) + ci * 16,
                    A + (size_t)(m0 + arow) * K + k0 + c * 8);
    }
  };
  auto fetch_b = [&](int k0, BF8 t[4]) {
    const __bf16* g = B + (size_t)(k0 + bk) * N + n0 + boff;
#pragma unroll
    for (int i = 0; i < 4; i++) t[i].v = *(const v8bf*)(g + i * 8);
  };
  auto store_b = [&](BF8 t[4], int p) {
#pragma unroll
    for (int i = 0; i < 4; i++)
#pragma unroll
      for (int e = 0; e < 8; e++)
        Bsm[p][(boff + i * 8 + e) * 64 + bk] = t[i].e[e];
  };

  F8 acc[2][4];
#pragma unroll
  for (int i = 0; i < 2; i++)
#pragma unroll
    for (int j = 0; j < 4; j++)
#pragma unroll
      for (int e = 0; e < 8; e++) acc[i][j].f[e] = 0.f;

  // ---- pipeline prologue: stage tile 0 into buffer 0
  {
    stage_a(0, 0);
    BF8 t0[4];
    fetch_b(0, t0);
    store_b(t0, 0);
    wait_async0();
  }
  __syncthreads();

  const int nt = K / 64;
  for (int it = 0; it < nt; ++it) {
    const int p    = it & 1;
    const int k0n  = (it + 1) * 64;
    const bool more = (it + 1) < nt;

    BF8 tn[4];
    if (more) {
      // barrier of previous iteration guarantees nobody still reads buf 1-p
      stage_a(k0n, 1 - p);
      fetch_b(k0n, tn);
      // warm L2 two tiles ahead for the DMA engine
      if (it + 2 < nt) {
        __builtin_prefetch(A + (size_t)(m0 + arow) * K + (it + 2) * 64, 0, 1);
        __builtin_prefetch(B + (size_t)((it + 2) * 64 + bk) * N + n0 + boff, 0, 1);
      }
    }

    // ---- compute tile `it` from buffer p
#pragma unroll
    for (int kk = 0; kk < 64; kk += 32) {
      v16bf af[2], bf[4];
#pragma unroll
      for (int mi = 0; mi < 2; mi++)
        af[mi] = load_frag_a(&Asm[p][(wm * 32 + mi * 16) * 64 + kk], 64, lane);
#pragma unroll
      for (int ni = 0; ni < 4; ni++)
        bf[ni] = load_frag_b(&Bsm[p][(wn * 64 + ni * 16) * 64 + kk], 64, lane);
#pragma unroll
      for (int mi = 0; mi < 2; mi++)
#pragma unroll
        for (int ni = 0; ni < 4; ni++)
          acc[mi][ni].v = WMMA_BF16(af[mi], bf[ni], acc[mi][ni].v);
    }

    if (more) store_b(tn, 1 - p);
    wait_async0();
    __syncthreads();
  }

  // ---- epilogue (C layout: VGPR r, lane -> row = r + 8*(lane>>4), col = lane&15)
  const int rbase = (lane >> 4) * 8;
  const int cl    = lane & 15;

  if (mode == 1) {
#pragma unroll
    for (int mi = 0; mi < 2; mi++)
#pragma unroll
      for (int ni = 0; ni < 4; ni++)
#pragma unroll
        for (int r = 0; r < 8; r++) {
          int row = m0 + wm * 32 + mi * 16 + r + rbase;
          int col = n0 + wn * 64 + ni * 16 + cl;
          Cf[(size_t)row * N + col] = acc[mi][ni].f[r];
        }
  } else {
    // wave's 64 columns span exactly one (part, head)
    const int c0   = n0 + wn * 64;
    const int part = c0 >> 10;            // 0=q 1=k 2=v
    const int h    = (c0 & 1023) >> 6;
#pragma unroll
    for (int mi = 0; mi < 2; mi++)
#pragma unroll
      for (int r = 0; r < 8; r++) {
        int gm = m0 + wm * 32 + mi * 16 + r + rbase;
        int b  = gm >> 11;                // S = 2048
        int s  = gm & 2047;
#pragma unroll
        for (int ni = 0; ni < 4; ni++) {
          int d = ni * 16 + cl;
          float val = acc[mi][ni].f[r];
          if (part == 2) {
            // V transposed: Vt[b,h,d,s]
            Vtb[(((size_t)(b * 16 + h) * 64 + d) << 11) + s] = (__bf16)val;
          } else {
            // RoPE pair (d, d+-32) lives in tile ni^2, same lane/VGPR
            float pv  = acc[mi][ni ^ 2].f[r];
            float rot = (d < 32) ? -pv : pv;
            float inv_freq = __expf(-(float)(d & 31) * 0.2878231366f); // ln(1e4)/32
            float ang = (float)s * inv_freq;
            float res = val * __cosf(ang) + rot * __sinf(ang);
            if (part == 0) res *= 0.125f;  // fold 64^-0.5 into q
            __bf16* dst = (part == 0) ? Qb : Kb;
            dst[(((size_t)(b * 16 + h) * 2048 + s) << 6) + d] = (__bf16)res;
          }
        }
      }
  }
}

// ---------------------------------------------------------------------------
// Flash attention: 4 waves/block share one (b,h); each wave owns a 16-row q
// tile.  K and Vt chunks (64x64 each) are DMA'd into double-buffered LDS so
// the next chunk streams in while the current one is consumed; P is staged
// per-wave through LDS to re-layout C-fragments into A-fragments.
__global__ __launch_bounds__(128) void attn_kernel(
    const __bf16* __restrict__ Qb, const __bf16* __restrict__ Kb,
    const __bf16* __restrict__ Vtb, __bf16* __restrict__ Ob)
{
  __shared__ __align__(16) __bf16 Ksm[2][64 * 64];   // [kv][d]   2x8KB
  __shared__ __align__(16) __bf16 Vsm[2][64 * 64];   // [d][kv]   2x8KB
  __shared__ __align__(16) __bf16 Psm[4][16 * 64];   // per-wave  8KB

  const int lane = threadIdx.x & 31;
  const int wid  = threadIdx.x >> 5;
  const int bh   = blockIdx.y;                      // 0..63
  const int b    = bh >> 4;
  const int h    = bh & 15;
  const int q0   = (blockIdx.x * 4 + wid) * 16;

  const __bf16* Qh = Qb  + (size_t)bh * 2048 * 64;
  const __bf16* Kh = Kb  + (size_t)bh * 2048 * 64;
  const __bf16* Vh = Vtb + (size_t)bh * 64 * 2048;
  __bf16* Ps = &Psm[wid][0];

  const unsigned k_base = lds_addr(&Ksm[0][0]);
  const unsigned v_base = lds_addr(&Vsm[0][0]);

  // stage one 64x64 K chunk and one 64x64 Vt chunk (512 x 16B each)
  auto stage_kv = [&](int j, int p) {
    const unsigned pb = (unsigned)p * (64 * 64 * 2);
#pragma unroll
    for (int c = 0; c < 4; c++) {
      int ci  = c * 128 + threadIdx.x;     // 0..511
      int row = ci >> 2;
      int seg = ci & 3;
      async_ld_b128(k_base + pb + ci * 16, Kh + (size_t)(j + row) * 64 + seg * 8);
      async_ld_b128(v_base + pb + ci * 16, Vh + (size_t)row * 2048 + j + seg * 8);
    }
  };

  v16bf qf[2];
#pragma unroll
  for (int c = 0; c < 2; c++)
    qf[c] = load_frag_a(Qh + (size_t)q0 * 64 + c * 32, 64, lane);

  F8 o[4];
  float m[8], l[8];
#pragma unroll
  for (int t = 0; t < 4; t++)
#pragma unroll
    for (int e = 0; e < 8; e++) o[t].f[e] = 0.f;
#pragma unroll
  for (int r = 0; r < 8; r++) { m[r] = -1e30f; l[r] = 0.f; }

  // ---- pipeline prologue
  stage_kv(0, 0);
  wait_async0();
  __syncthreads();

  for (int jt = 0; jt < 32; ++jt) {
    const int j = jt * 64;
    const int p = jt & 1;
    if (jt + 1 < 32) stage_kv(j + 64, 1 - p);   // stream next chunk during compute

    // S = Q K^T over a 16x64 strip (scale already folded into Q)
    F8 s4[4];
#pragma unroll
    for (int t = 0; t < 4; t++) {
#pragma unroll
      for (int e = 0; e < 8; e++) s4[t].f[e] = 0.f;
#pragma unroll
      for (int c = 0; c < 2; c++) {
        v16bf kf = load_frag_b(&Ksm[p][(t * 16) * 64 + c * 32], 64, lane);
        s4[t].v = WMMA_BF16(qf[c], kf, s4[t].v);
      }
    }
    // online softmax: row max across 4 tiles + 16-lane group
    float alpha[8];
#pragma unroll
    for (int r = 0; r < 8; r++) {
      float mx = s4[0].f[r];
#pragma unroll
      for (int t = 1; t < 4; t++) mx = fmaxf(mx, s4[t].f[r]);
#pragma unroll
      for (int off = 1; off < 16; off <<= 1) mx = fmaxf(mx, __shfl_xor(mx, off, 32));
      float mn = fmaxf(m[r], mx);
      alpha[r] = __expf(m[r] - mn);
      m[r] = mn;
    }
    float rs[8];
#pragma unroll
    for (int r = 0; r < 8; r++) rs[r] = 0.f;
#pragma unroll
    for (int t = 0; t < 4; t++)
#pragma unroll
      for (int r = 0; r < 8; r++) {
        float pp = __expf(s4[t].f[r] - m[r]);
        rs[r] += pp;
        Ps[(r + ((lane >> 4) * 8)) * 64 + t * 16 + (lane & 15)] = (__bf16)pp;
      }
#pragma unroll
    for (int r = 0; r < 8; r++) {
#pragma unroll
      for (int off = 1; off < 16; off <<= 1) rs[r] += __shfl_xor(rs[r], off, 32);
      l[r] = l[r] * alpha[r] + rs[r];
    }
#pragma unroll
    for (int t = 0; t < 4; t++)
#pragma unroll
      for (int r = 0; r < 8; r++) o[t].f[r] *= alpha[r];

    // O += P V  (P read back from LDS as A-fragments; LDS in-order per wave)
#pragma unroll
    for (int c = 0; c < 2; c++) {
      v16bf pf = load_frag_a(Ps + c * 32, 64, lane);
#pragma unroll
      for (int t = 0; t < 4; t++) {
        v16bf vf = load_frag_b(&Vsm[p][(t * 16) * 64 + c * 32], 64, lane);
        o[t].v = WMMA_BF16(pf, vf, o[t].v);
      }
    }

    wait_async0();
    __syncthreads();
  }

  // normalize + store to Ob[B, S, H*64]
#pragma unroll
  for (int r = 0; r < 8; r++) {
    float inv = 1.f / l[r];
    int row = q0 + r + (lane >> 4) * 8;
#pragma unroll
    for (int t = 0; t < 4; t++) {
      int col = h * 64 + t * 16 + (lane & 15);
      Ob[((size_t)b * 2048 + row) * 1024 + col] = (__bf16)(o[t].f[r] * inv);
    }
  }
}

// ---------------------------------------------------------------------------
extern "C" void kernel_launch(void* const* d_in, const int* in_sizes, int n_in,
                              void* d_out, int out_size, void* d_ws, size_t ws_size,
                              hipStream_t stream) {
  (void)in_sizes; (void)n_in; (void)out_size; (void)ws_size;
  const float* x    = (const float*)d_in[0];
  const float* Wqkv = (const float*)d_in[1];
  const float* Wout = (const float*)d_in[2];
  float* out = (float*)d_out;

  const int MS = 4 * 2048;  // B*S = 8192

  size_t off = 0;
  char* ws = (char*)d_ws;
  auto alloc = [&](size_t bytes) -> void* {
    void* p = ws + off;
    off += (bytes + 255) & ~(size_t)255;
    return p;
  };
  __bf16* xb     = (__bf16*)alloc((size_t)MS * 1024 * 2);
  __bf16* wqkvb  = (__bf16*)alloc((size_t)1024 * 3072 * 2);
  __bf16* woutb  = (__bf16*)alloc((size_t)1024 * 1024 * 2);
  __bf16* Qb     = (__bf16*)alloc((size_t)64 * 2048 * 64 * 2);
  __bf16* Kb     = (__bf16*)alloc((size_t)64 * 2048 * 64 * 2);
  __bf16* Vtb    = (__bf16*)alloc((size_t)64 * 64 * 2048 * 2);
  __bf16* Ob     = (__bf16*)alloc((size_t)MS * 1024 * 2);

  cvt_f32_bf16<<<dim3((MS * 1024) / 8 / 256), 256, 0, stream>>>(x, xb, MS * 1024);
  cvt_f32_bf16<<<dim3((1024 * 3072) / 8 / 256), 256, 0, stream>>>(Wqkv, wqkvb, 1024 * 3072);
  cvt_f32_bf16<<<dim3((1024 * 1024) / 8 / 256), 256, 0, stream>>>(Wout, woutb, 1024 * 1024);

  // QKV projection + fused RoPE/scatter
  gemm_bf16_kernel<<<dim3(MS / 128, 3072 / 128), 256, 0, stream>>>(
      xb, wqkvb, MS, 3072, 1024, /*mode=*/0, nullptr, Qb, Kb, Vtb);

  // flash attention
  attn_kernel<<<dim3(2048 / 64, 64), 128, 0, stream>>>(Qb, Kb, Vtb, Ob);

  // output projection -> fp32
  gemm_bf16_kernel<<<dim3(MS / 128, 1024 / 128), 256, 0, stream>>>(
      Ob, woutb, MS, 1024, 1024, /*mode=*/1, out, nullptr, nullptr, nullptr);
}
